// Encoder_conv_55035710931026
// MI455X (gfx1250) — compile-verified, matrix operand order checked by reading, not run on hardware
//
#include <hip/hip_runtime.h>
#include <hip/hip_bf16.h>
#include <math.h>

// ---------- types ----------
typedef __attribute__((ext_vector_type(16))) __bf16 v16bf;
typedef __attribute__((ext_vector_type(8)))  float  v8f;

__device__ __forceinline__ float bf2f(unsigned short u) {
    unsigned int x = ((unsigned int)u) << 16;
    float f; __builtin_memcpy(&f, &x, 4); return f;
}
__device__ __forceinline__ unsigned short f2bf(float f) {
    unsigned int x; __builtin_memcpy(&x, &f, 4);
    unsigned int r = x + 0x7FFFu + ((x >> 16) & 1u);  // round-to-nearest-even
    return (unsigned short)(r >> 16);
}

// ---------- workspace layout (bytes, all 256-aligned) ----------
#define OFF_H0   ((size_t)0)                     // [8,256,256,128] bf16 = 134217728
#define OFF_H1   (OFF_H0 + 134217728ull)         // [8,64,64,256]  bf16 = 16777216
#define OFF_H2   (OFF_H1 + 16777216ull)          // [8,16,16,256]  bf16 = 1048576
#define OFF_H3   (OFF_H2 + 1048576ull)           // [8,8,8,256]    bf16 = 262144
#define OFF_T1   (OFF_H3 + 262144ull)            // [512,1024]     bf16 = 1048576
#define OFF_HR1  (OFF_T1 + 1048576ull)           // [512,256]      bf16 = 262144
#define OFF_HR2  (OFF_HR1 + 262144ull)           // [512,256]      bf16 = 262144
#define OFF_PW1  (OFF_HR2 + 262144ull)           // 256*6272  bf16 = 3211264
#define OFF_PW2  (OFF_PW1 + 3211264ull)          // 256*12544 bf16 = 6422528
#define OFF_PW3  (OFF_PW2 + 6422528ull)          // 256*2304  bf16 = 1179648
#define OFF_PR0A (OFF_PW3 + 1179648ull)          // 1024*768  bf16 = 1572864
#define OFF_PR0B (OFF_PR0A + 1572864ull)         // 256*1024  bf16 = 524288
#define OFF_PR1A (OFF_PR0B + 524288ull)          // 1572864
#define OFF_PR1B (OFF_PR1A + 1572864ull)         // 524288
#define OFF_IDX  (OFF_PR1B + 524288ull)          // 512*4 = 2048
#define OFF_HIST (OFF_IDX + 2048ull)             // 512*4 = 2048
#define OFF_MSE  (OFF_HIST + 2048ull)            // 256

// ---------- weight repack: OIHW f32 -> [Cout][KH][KW][Cin] bf16 ----------
__global__ void k_pack_w(const float* __restrict__ src, unsigned short* __restrict__ dst,
                         int Cout, int Cin, int KH, int KW) {
    int total = Cout * Cin * KH * KW;
    int i = blockIdx.x * 256 + threadIdx.x;
    if (i >= total) return;
    int ci = i % Cin;  int t = i / Cin;
    int kw = t % KW;   t /= KW;
    int kh = t % KH;   int co = t / KH;
    float v = src[((co * Cin + ci) * KH + kh) * KW + kw];
    dst[i] = f2bf(v);  // dst index == ((co*KH+kh)*KW+kw)*Cin + ci == i by construction
}

// ---------- conv_in: [8,3,512,512] f32 -> relu -> [8,256,256,128] NHWC bf16 ----------
__global__ __launch_bounds__(256)
void k_conv_in(const float* __restrict__ x, const float* __restrict__ w,
               const float* __restrict__ bias, unsigned short* __restrict__ out) {
    int tid = blockIdx.x * 256 + threadIdx.x;       // 8*256*256*16 threads
    int cg  = tid & 15;                             // 8-channel group
    int pix = tid >> 4;
    int ow = pix & 255; int t = pix >> 8;
    int oh = t & 255;   int b = t >> 8;
    float acc[8];
    #pragma unroll
    for (int j = 0; j < 8; ++j) acc[j] = bias[cg * 8 + j];
    #pragma unroll
    for (int kh = 0; kh < 4; ++kh) {
        int ih = oh * 2 - 1 + kh;
        if ((unsigned)ih >= 512u) continue;
        #pragma unroll
        for (int kw = 0; kw < 4; ++kw) {
            int iw = ow * 2 - 1 + kw;
            if ((unsigned)iw >= 512u) continue;
            #pragma unroll
            for (int ci = 0; ci < 3; ++ci) {
                float xv = x[((size_t)(b * 3 + ci) * 512 + ih) * 512 + iw];
                int wb = ci * 16 + kh * 4 + kw;
                #pragma unroll
                for (int j = 0; j < 8; ++j)
                    acc[j] += xv * w[(cg * 8 + j) * 48 + wb];
            }
        }
    }
    size_t o = (((size_t)(b * 256 + oh) * 256) + ow) * 128 + cg * 8;
    union { uint4 q; unsigned short s[8]; } pk;
    #pragma unroll
    for (int j = 0; j < 8; ++j) pk.s[j] = f2bf(fmaxf(acc[j], 0.0f));
    *(uint4*)(out + o) = pk.q;   // 16B-aligned: o is a multiple of 8 elements
}

// ---------- generic implicit-GEMM conv via bf16 WMMA ----------
// Each wave computes a 32(M) x 16(N) output tile: the gathered B fragment is
// reused by two WMMAs; K loop unrolled 2x (4 WMMAs/iter) for ILP.
// in:  NHWC bf16 [B,Hin,Win,Cin]       wp: bf16 [Cout][KH*KW*Cin] (kh,kw,cin order)
// out: NHWC bf16 [B,Hout,Wout,Cout]    res: optional NHWC bf16 residual, bias optional f32
// Requirements: Cin % 32 == 0, Ktot % 64 == 0, (B*Hout*Wout) % 16 == 0, Cout % 128 == 0
__global__ __launch_bounds__(128)
void k_conv_wmma(const unsigned short* __restrict__ in,
                 const unsigned short* __restrict__ wp,
                 const float* __restrict__ bias,
                 const unsigned short* __restrict__ res,
                 unsigned short* __restrict__ out,
                 int Hin, int Win, int Cin, int Hout, int Wout, int Cout,
                 int KH, int KW, int stride, int padH, int padW,
                 int relu_in, int relu_out) {
    const int lane = threadIdx.x & 31;
    const int wv   = threadIdx.x >> 5;                  // 0..3
    const int n0   = blockIdx.x * 16;                   // N tile (spatial)
    const int m0   = (blockIdx.y * 4 + wv) * 32;        // 32-wide M tile (out channels)
    const int Ktot = KH * KW * Cin;

    // column decode (once, outside the K loop)
    const int n  = n0 + (lane & 15);
    const int ow = n % Wout;
    const int nt = n / Wout;
    const int oh = nt % Hout;
    const int b  = nt / Hout;
    const int ohs = oh * stride - padH;                 // input row base
    const int ows = ow * stride - padW;                 // input col base
    const int khalf = (lane >> 4) << 3;                 // 0 or 8 (ISA K-half per lane group)

    const unsigned short* wrow0 = wp + (size_t)(m0 + (lane & 15)) * Ktot + khalf;
    const unsigned short* wrow1 = wrow0 + (size_t)16 * Ktot;

    v8f acc0 = {0.f, 0.f, 0.f, 0.f, 0.f, 0.f, 0.f, 0.f};
    v8f acc1 = {0.f, 0.f, 0.f, 0.f, 0.f, 0.f, 0.f, 0.f};

    // incremental (kh,kw,ci) bookkeeping: Cin % 32 == 0, so a 32-wide K step
    // never crosses a (kh,kw) boundary -> single gather base per sub-step.
    int ci = 0, kw = 0, kh = 0;
    for (int k = 0; k < Ktot; k += 64) {
        // prefetch weight stream two iterations ahead (speculative OK)
        __builtin_prefetch(wrow0 + k + 128, 0, 1);
        __builtin_prefetch(wrow1 + k + 128, 0, 1);
        #pragma unroll
        for (int u = 0; u < 2; ++u) {
            const int koff = k + u * 32;
            const int ih = ohs + kh;
            const int iw = ows + kw;

            union { v16bf v; uint4 q[2]; unsigned short s[16]; } Bm;
            uint4 d0 = {0u, 0u, 0u, 0u}, d1 = {0u, 0u, 0u, 0u};
            if ((unsigned)ih < (unsigned)Hin && (unsigned)iw < (unsigned)Win) {
                const unsigned short* src =
                    in + ((size_t)(b * Hin + ih) * Win + iw) * Cin + ci + khalf;
                d0 = *(const uint4*)(src);
                d1 = *(const uint4*)(src + 16);
                __builtin_prefetch(src + 32, 0, 1);   // next channel chunk
            }
            Bm.q[0] = d0; Bm.q[1] = d1;
            if (relu_in) {
                #pragma unroll
                for (int j = 0; j < 16; ++j)
                    Bm.s[j] = (Bm.s[j] & 0x8000u) ? (unsigned short)0 : Bm.s[j];
            }

            union { v16bf v; uint4 q[2]; } A0, A1;
            A0.q[0] = *(const uint4*)(wrow0 + koff);
            A0.q[1] = *(const uint4*)(wrow0 + koff + 16);
            A1.q[0] = *(const uint4*)(wrow1 + koff);
            A1.q[1] = *(const uint4*)(wrow1 + koff + 16);

            acc0 = __builtin_amdgcn_wmma_f32_16x16x32_bf16(false, A0.v, false, Bm.v,
                                                           (short)0, acc0, false, false);
            acc1 = __builtin_amdgcn_wmma_f32_16x16x32_bf16(false, A1.v, false, Bm.v,
                                                           (short)0, acc1, false, false);

            ci += 32;
            if (ci == Cin) { ci = 0; ++kw; if (kw == KW) { kw = 0; ++kh; } }
        }
    }

    // packed epilogue: each tile's 8 channels are contiguous -> b128 load/store
    const size_t obase = (size_t)n * Cout;
    const int cA = m0 + khalf;                           // tile0 channels cA..cA+7
    union F8 { float4 v[2]; float f[8]; };
    F8 b0, b1;
    if (bias) {
        const float4* bp = (const float4*)(bias + cA);
        b0.v[0] = bp[0]; b0.v[1] = bp[1];                // 32B-aligned (cA % 8 == 0)
        const float4* bq = (const float4*)(bias + cA + 16);
        b1.v[0] = bq[0]; b1.v[1] = bq[1];
    }
    union U8 { uint4 q; unsigned short s[8]; };
    U8 r0u, r1u;
    if (res) {
        r0u.q = *(const uint4*)(res + obase + cA);
        r1u.q = *(const uint4*)(res + obase + cA + 16);
    }
    U8 s0, s1;
    #pragma unroll
    for (int j = 0; j < 8; ++j) {
        float v0 = acc0[j];
        float v1 = acc1[j];
        if (bias) { v0 += b0.f[j]; v1 += b1.f[j]; }
        if (res)  { v0 += bf2f(r0u.s[j]); v1 += bf2f(r1u.s[j]); }
        if (relu_out) { v0 = fmaxf(v0, 0.0f); v1 = fmaxf(v1, 0.0f); }
        s0.s[j] = f2bf(v0);
        s1.s[j] = f2bf(v1);
    }
    *(uint4*)(out + obase + cA)      = s0.q;
    *(uint4*)(out + obase + cA + 16) = s1.q;
}

// ---------- VQ: argmin over 512 codes; one wave per latent vector ----------
__global__ __launch_bounds__(32)
void k_vq_argmin(const unsigned short* __restrict__ lat,  // bf16 [512][256] (pre-relu)
                 const float* __restrict__ emb,           // f32 [512][256]
                 int* __restrict__ idx, int* __restrict__ hist) {
    int n = blockIdx.x;
    int lane = threadIdx.x;
    float lv[8];
    #pragma unroll
    for (int j = 0; j < 8; ++j)
        lv[j] = fmaxf(bf2f(lat[(size_t)n * 256 + lane * 8 + j]), 0.0f);
    float best = 3.4e38f; int bi = 0;
    for (int c = 0; c < 512; ++c) {
        const float* e = emb + (size_t)c * 256 + lane * 8;
        float p = 0.f;
        #pragma unroll
        for (int j = 0; j < 8; ++j) { float ev = e[j]; p += ev * ev - 2.0f * lv[j] * ev; }
        #pragma unroll
        for (int o = 16; o > 0; o >>= 1) p += __shfl_xor(p, o, 32);
        if (p < best) { best = p; bi = c; }   // strict <  => first index on ties
    }
    if (lane == 0) { idx[n] = bi; atomicAdd(&hist[bi], 1); }
}

// ---------- VQ: scatter quantized output (NCHW f32) + MSE accumulation ----------
__global__ __launch_bounds__(256)
void k_vq_out(const float* __restrict__ emb, const int* __restrict__ idx,
              const unsigned short* __restrict__ lat, float* __restrict__ outq,
              float* __restrict__ mse) {
    __shared__ float red[256];
    int tid = blockIdx.x * 256 + threadIdx.x;            // 16384 total
    int ow = tid & 7; int t = tid >> 3;
    int oh = t & 7;   t >>= 3;
    int c  = t & 255; int b = t >> 8;
    int n  = (b * 8 + oh) * 8 + ow;
    float q = emb[(size_t)idx[n] * 256 + c];
    outq[tid] = q;                                        // quant_st value == quant
    float l = fmaxf(bf2f(lat[(size_t)n * 256 + c]), 0.0f);
    float d = q - l;
    red[threadIdx.x] = d * d;
    __syncthreads();
    for (int s = 128; s > 0; s >>= 1) {
        if (threadIdx.x < s) red[threadIdx.x] += red[threadIdx.x + s];
        __syncthreads();
    }
    if (threadIdx.x == 0) atomicAdd(mse, red[0]);
}

// ---------- VQ: loss + perplexity ----------
__global__ __launch_bounds__(512)
void k_vq_finalize(const int* __restrict__ hist, const float* __restrict__ mse,
                   float* __restrict__ out_loss, float* __restrict__ out_ppl) {
    __shared__ float red[512];
    float p = (float)hist[threadIdx.x] / 512.0f;
    red[threadIdx.x] = p * logf(p + 1e-10f);
    __syncthreads();
    for (int s = 256; s > 0; s >>= 1) {
        if (threadIdx.x < s) red[threadIdx.x] += red[threadIdx.x + s];
        __syncthreads();
    }
    if (threadIdx.x == 0) {
        *out_loss = 2.0f * (*mse) / (512.0f * 256.0f);   // q_loss + BETA*e_loss, BETA=1
        *out_ppl  = expf(-red[0]);
    }
}

// ---------- launcher ----------
extern "C" void kernel_launch(void* const* d_in, const int* in_sizes, int n_in,
                              void* d_out, int out_size, void* d_ws, size_t ws_size,
                              hipStream_t stream) {
    const float* x     = (const float*)d_in[0];
    const float* w_in  = (const float*)d_in[1];
    const float* b_in  = (const float*)d_in[2];
    const float* w_h1  = (const float*)d_in[3];
    const float* b_h1  = (const float*)d_in[4];
    const float* w_h2  = (const float*)d_in[5];
    const float* b_h2  = (const float*)d_in[6];
    const float* w_h3  = (const float*)d_in[7];
    const float* b_h3  = (const float*)d_in[8];
    const float* r0w1  = (const float*)d_in[9];
    const float* r0w2  = (const float*)d_in[10];
    const float* r1w1  = (const float*)d_in[11];
    const float* r1w2  = (const float*)d_in[12];
    const float* emb   = (const float*)d_in[13];

    char* ws = (char*)d_ws;
    unsigned short* h0  = (unsigned short*)(ws + OFF_H0);
    unsigned short* h1  = (unsigned short*)(ws + OFF_H1);
    unsigned short* h2  = (unsigned short*)(ws + OFF_H2);
    unsigned short* h3  = (unsigned short*)(ws + OFF_H3);
    unsigned short* t1  = (unsigned short*)(ws + OFF_T1);
    unsigned short* hr1 = (unsigned short*)(ws + OFF_HR1);
    unsigned short* hr2 = (unsigned short*)(ws + OFF_HR2);
    unsigned short* pw1  = (unsigned short*)(ws + OFF_PW1);
    unsigned short* pw2  = (unsigned short*)(ws + OFF_PW2);
    unsigned short* pw3  = (unsigned short*)(ws + OFF_PW3);
    unsigned short* pr0a = (unsigned short*)(ws + OFF_PR0A);
    unsigned short* pr0b = (unsigned short*)(ws + OFF_PR0B);
    unsigned short* pr1a = (unsigned short*)(ws + OFF_PR1A);
    unsigned short* pr1b = (unsigned short*)(ws + OFF_PR1B);
    int*   idxp  = (int*)(ws + OFF_IDX);
    int*   histp = (int*)(ws + OFF_HIST);
    float* msep  = (float*)(ws + OFF_MSE);

    // zero histogram + mse accumulator (memset node is graph-capturable)
    hipMemsetAsync(ws + OFF_HIST, 0, 2048 + 256, stream);

    // repack weights to [Cout][KH][KW][Cin] bf16
    auto packLaunch = [&](const float* src, unsigned short* dst, int Co, int Ci, int KH, int KW) {
        int total = Co * Ci * KH * KW;
        k_pack_w<<<dim3((total + 255) / 256), 256, 0, stream>>>(src, dst, Co, Ci, KH, KW);
    };
    packLaunch(w_h1, pw1, 256, 128, 7, 7);
    packLaunch(w_h2, pw2, 256, 256, 7, 7);
    packLaunch(w_h3, pw3, 256, 256, 3, 3);
    packLaunch(r0w1, pr0a, 1024, 256, 1, 3);
    packLaunch(r0w2, pr0b, 256, 1024, 1, 1);
    packLaunch(r1w1, pr1a, 1024, 256, 1, 3);
    packLaunch(r1w2, pr1b, 256, 1024, 1, 1);

    // conv_in (direct, f32 -> relu -> bf16 NHWC)
    k_conv_in<<<dim3(32768), 256, 0, stream>>>(x, w_in, b_in, h0);

    // conv_h1: [8,256,256,128] -> [8,64,64,256], 7x7 s4 p2, relu
    k_conv_wmma<<<dim3(32768 / 16, 256 / 128), 128, 0, stream>>>(
        h0, pw1, b_h1, nullptr, h1, 256, 256, 128, 64, 64, 256, 7, 7, 4, 2, 2, 0, 1);
    // conv_h2: -> [8,16,16,256], 7x7 s4 p2, relu
    k_conv_wmma<<<dim3(2048 / 16, 256 / 128), 128, 0, stream>>>(
        h1, pw2, b_h2, nullptr, h2, 64, 64, 256, 16, 16, 256, 7, 7, 4, 2, 2, 0, 1);
    // conv_h3: -> [8,8,8,256], 3x3 s2 p1, relu  => H
    k_conv_wmma<<<dim3(512 / 16, 256 / 128), 128, 0, stream>>>(
        h2, pw3, b_h3, nullptr, h3, 16, 16, 256, 8, 8, 256, 3, 3, 2, 1, 1, 0, 1);

    // residual block 0: t = conv1x3(relu(H)) -> relu -> conv1x1 -> H1 = H + t
    k_conv_wmma<<<dim3(512 / 16, 1024 / 128), 128, 0, stream>>>(
        h3, pr0a, nullptr, nullptr, t1, 8, 8, 256, 8, 8, 1024, 1, 3, 1, 0, 1, 1, 1);
    k_conv_wmma<<<dim3(512 / 16, 256 / 128), 128, 0, stream>>>(
        t1, pr0b, nullptr, h3, hr1, 8, 8, 1024, 8, 8, 256, 1, 1, 1, 0, 0, 0, 0);
    // residual block 1: H2 = H1 + conv1x1(relu(conv1x3(relu(H1))))
    k_conv_wmma<<<dim3(512 / 16, 1024 / 128), 128, 0, stream>>>(
        hr1, pr1a, nullptr, nullptr, t1, 8, 8, 256, 8, 8, 1024, 1, 3, 1, 0, 1, 1, 1);
    k_conv_wmma<<<dim3(512 / 16, 256 / 128), 128, 0, stream>>>(
        t1, pr1b, nullptr, hr1, hr2, 8, 8, 1024, 8, 8, 256, 1, 1, 1, 0, 0, 0, 0);

    // vector quantizer (final relu fused into lat reads)
    k_vq_argmin<<<dim3(512), 32, 0, stream>>>(hr2, emb, idxp, histp);
    k_vq_out<<<dim3(64), 256, 0, stream>>>(emb, idxp, hr2, (float*)d_out, msep);
    k_vq_finalize<<<dim3(1), 512, 0, stream>>>(histp, msep,
                                               (float*)d_out + 16384, (float*)d_out + 16385);

    (void)in_sizes; (void)n_in; (void)out_size; (void)ws_size;
}
